// YoloV9Trainer_17411797418697
// MI455X (gfx1250) — matrix operand order repaired and checked by reading, not compiled, not run on hardware
//
#include <hip/hip_runtime.h>
#include <math.h>

// ---------------- problem constants ----------------
constexpr int kB   = 2;
constexpr int kM   = 48;
constexpr int kA   = 5376;           // 64*64 + 32*32 + 16*16
constexpr int kC   = 80;
constexpr int kE   = 32;
constexpr int kHW  = 48 * 48;        // 2304
constexpr int kTOPK = 10;
constexpr float kEPS = 1e-9f;
// per-batch output layout: [A*85 box_target | A*HW pred_masks | A*HW tgt_masks]
constexpr size_t kOutB = (size_t)kA * (85 + 2 * kHW);   // 25,229,568

typedef _Float16 v16h __attribute__((ext_vector_type(16)));
typedef float    v8f  __attribute__((ext_vector_type(8)));

// ============================================================
// Kernel 1: per-(b,m) row — target matrix, row max_t / max_iou,
//           10th-largest threshold.  One block per (b,m).
//           Top-k via per-thread sorted top-10 lists + parallel
//           pairwise tree merge (log2(256)=8 levels).
// ============================================================
__global__ void __launch_bounds__(256)
k1_assign(const float* __restrict__ cls_logits,   // (B,A,C)
          const float* __restrict__ pred_boxes,   // (B,A,4)
          const float* __restrict__ true_boxes,   // (B,M,4)
          const float* __restrict__ anchors,      // (A,2)
          const int*   __restrict__ true_cls,     // (B,M)
          float* __restrict__ target,             // ws: (B*M, A)
          float* __restrict__ rowstats)           // ws: (B*M, 3) = max_t, max_iou, thresh
{
    const int bm = blockIdx.x;          // 0 .. B*M-1
    const int b  = bm / kM;
    const int tid = threadIdx.x;

    const float gx1 = true_boxes[(size_t)bm * 4 + 0];
    const float gy1 = true_boxes[(size_t)bm * 4 + 1];
    const float gx2 = true_boxes[(size_t)bm * 4 + 2];
    const float gy2 = true_boxes[(size_t)bm * 4 + 3];
    const float w1 = gx2 - gx1, h1 = gy2 - gy1;
    const float atan_g = atanf(w1 / (h1 + kEPS));
    const float inv_pi2 = 4.0f / (float)(M_PI * M_PI);
    const int ci = true_cls[bm];

    float lmaxT = 0.f, lmaxI = 0.f;
    float top[kTOPK];
#pragma unroll
    for (int i = 0; i < kTOPK; ++i) top[i] = 0.f;

    for (int a = tid; a < kA; a += 256) {
        const float ax = anchors[2 * a + 0];
        const float ay = anchors[2 * a + 1];
        const bool gm = (ax > gx1) & (ay > gy1) & (ax < gx2) & (ay < gy2);

        const float* pb = pred_boxes + ((size_t)b * kA + a) * 4;
        const float px1 = pb[0], py1 = pb[1], px2 = pb[2], py2 = pb[3];

        // ---- CIoU (reference order) ----
        const float ix1 = fmaxf(gx1, px1), iy1 = fmaxf(gy1, py1);
        const float ix2 = fminf(gx2, px2), iy2 = fminf(gy2, py2);
        const float inter = fmaxf(ix2 - ix1, 0.f) * fmaxf(iy2 - iy1, 0.f);
        const float w2 = px2 - px1, h2 = py2 - py1;
        const float uni = w1 * h1 + w2 * h2 - inter;
        const float iou = inter / (uni + kEPS);
        const float cw = fmaxf(gx2, px2) - fminf(gx1, px1);
        const float ch = fmaxf(gy2, py2) - fminf(gy1, py1);
        const float c2 = cw * cw + ch * ch + kEPS;
        const float dx = gx1 + gx2 - px1 - px2;
        const float dy = gy1 + gy2 - py1 - py2;
        const float rho2 = (dx * dx + dy * dy) * 0.25f;
        const float dat = atan_g - atanf(w2 / (h2 + kEPS));
        const float v = inv_pi2 * dat * dat;
        const float alpha = v / (1.0f - iou + v + kEPS);
        const float ciou = iou - rho2 / c2 - alpha * v;
        const float iou_c = fmaxf(ciou, 0.f);

        float t = 0.f;
        if (gm) {
            const float lg = cls_logits[((size_t)b * kA + a) * kC + ci];
            const float p = 1.0f / (1.0f + expf(-lg));
            const float i2 = iou_c * iou_c;
            t = (i2 * i2 * i2) * sqrtf(p);     // iou^6 * cls^0.5
        }
        target[(size_t)bm * kA + a] = t;
        lmaxT = fmaxf(lmaxT, t);
        lmaxI = fmaxf(lmaxI, iou_c);

        if (t > top[kTOPK - 1]) {              // sorted-descending insert
            int j = kTOPK - 1;
            while (j > 0 && top[j - 1] < t) { top[j] = top[j - 1]; --j; }
            top[j] = t;
        }
    }

    __shared__ float sT[256], sI[256];
    __shared__ float sTop[256 * kTOPK];
    sT[tid] = lmaxT;
    sI[tid] = lmaxI;
#pragma unroll
    for (int i = 0; i < kTOPK; ++i) sTop[tid * kTOPK + i] = top[i];
    __syncthreads();

    // parallel tree reduction: maxes + sorted-top10 pairwise merge
    for (int s = 128; s > 0; s >>= 1) {
        if (tid < s) {
            sT[tid] = fmaxf(sT[tid], sT[tid + s]);
            sI[tid] = fmaxf(sI[tid], sI[tid + s]);

            float av[kTOPK], bv[kTOPK], rv[kTOPK];
#pragma unroll
            for (int i = 0; i < kTOPK; ++i) {
                av[i] = sTop[tid * kTOPK + i];
                bv[i] = sTop[(tid + s) * kTOPK + i];
            }
            int ia = 0, ib = 0;
#pragma unroll
            for (int i = 0; i < kTOPK; ++i) {   // two-pointer merge, keep top-10
                const float va = av[ia], vb = bv[ib];
                if (va >= vb) { rv[i] = va; ++ia; }
                else          { rv[i] = vb; ++ib; }
            }
#pragma unroll
            for (int i = 0; i < kTOPK; ++i) sTop[tid * kTOPK + i] = rv[i];
        }
        __syncthreads();
    }

    if (tid == 0) {
        rowstats[bm * 3 + 0] = sT[0];
        rowstats[bm * 3 + 1] = sI[0];
        rowstats[bm * 3 + 2] = sTop[kTOPK - 1];   // 10th largest
    }
}

// ============================================================
// Kernel 2: per-(b,a) — argmax over m of thresholded target,
//           write 85-float box_target row + unique_idx.
// ============================================================
__global__ void __launch_bounds__(256)
k2_box(const float* __restrict__ true_boxes,
       const float* __restrict__ scalers,
       const int*   __restrict__ true_cls,
       const float* __restrict__ target,
       const float* __restrict__ rowstats,
       int*   __restrict__ uidx,
       float* __restrict__ out)
{
    const int idx = blockIdx.x * 256 + threadIdx.x;
    if (idx >= kB * kA) return;
    const int b = idx / kA;
    const int a = idx - b * kA;

    int best_m = 0;
    float best_v = 0.f;
    for (int m = 0; m < kM; ++m) {
        const int bm = b * kM + m;
        const float t  = target[(size_t)bm * kA + a];
        const float th = rowstats[bm * 3 + 2];
        const float tv = (t > 0.f && t >= th) ? t : 0.f;   // top-k scatter -> threshold keep
        if (tv > best_v) { best_v = tv; best_m = m; }      // first-occurrence argmax
    }
    const int bm = b * kM + best_m;
    const float vmf = (best_v > 0.f) ? 1.f : 0.f;
    uidx[idx] = best_m;

    const float* tb = true_boxes + (size_t)bm * 4;
    const float s = scalers[a];
    const float t_at = target[(size_t)bm * kA + a];
    const float nrm = t_at / (rowstats[bm * 3 + 0] + kEPS) * rowstats[bm * 3 + 1];
    const int cls = true_cls[bm];
    const float cv = nrm * vmf;

    float* o = out + (size_t)b * kOutB + (size_t)a * 85;
    o[0] = tb[0] / s;
    o[1] = tb[1] / s;
    o[2] = tb[2] / s;
    o[3] = tb[3] / s;
    o[4] = vmf;
#pragma unroll 4
    for (int c = 0; c < kC; ++c) o[5 + c] = (c == cls) ? cv : 0.f;
}

// ============================================================
// Kernel 3: pred_masks = (A x E) @ (E x HW) per batch via
//           V_WMMA_F32_16X16X32_F16 (K = E = 32 -> one wmma per
//           16x16 tile).  Block = one (b, col-tile, split); the
//           protos tile is staged f32->f16 in LDS once, B frag
//           held in registers, each wave sweeps 7 row tiles.
// ============================================================
constexpr int kCT    = kHW / 16;               // 144 col tiles
constexpr int kRT    = kA / 16;                // 336 row tiles
constexpr int kSplit = 6;                      // row-range splits per col tile
constexpr int kRTperWave = kRT / (8 * kSplit); // 7 wmma iterations per wave

__global__ void __launch_bounds__(256)
k3_wmma(const float* __restrict__ emb,   // (B,A,E)
        const float* __restrict__ pro,   // (B,HW,E)
        float* __restrict__ out)
{
    __shared__ _Float16 sB[16 * kE];     // one protos tile: 16 cols x 32 k

    const int tid  = threadIdx.x;
    const int wave = tid >> 5;
    const int lane = tid & 31;

    const int blk = blockIdx.x;
    const int b   = blk / (kCT * kSplit);
    const int rem = blk - b * (kCT * kSplit);
    const int ct  = rem / kSplit;
    const int sp  = rem - ct * kSplit;
    const int col0 = ct * 16;

    // cooperative stage of protos tile (contiguous 512 floats) -> f16 LDS
    {
        const float* pb = pro + ((size_t)b * kHW + col0) * kE;
        const int i0 = tid * 2;
        sB[i0]     = (_Float16)pb[i0];
        sB[i0 + 1] = (_Float16)pb[i0 + 1];
    }
    __syncthreads();

    const int n  = lane & 15;
    const int hi = lane >> 4;            // half-wave select

    // B fragment (32x16): K split across half-waves, N = lane&15.
    // Held in registers for all 7 row tiles.
    v16h bf;
    {
        const int kb = hi * 8;
#pragma unroll
        for (int i = 0; i < 16; ++i) bf[i] = sB[n * kE + kb + i + (i >= 8 ? 8 : 0)];
    }

    const int k1 = hi * 8;               // A-frag K offsets (16-bit A 16x32 layout)
    const int k2 = 16 + hi * 8;
    const int mb = hi * 8;               // D-layout row base
    const int rt0 = (sp * 8 + wave) * kRTperWave;

    const float* ebase = emb + (size_t)b * kA * kE;
    float* obase = out + (size_t)b * kOutB + (size_t)kA * 85 + col0;

#pragma unroll
    for (int it = 0; it < kRTperWave; ++it) {
        const int row0 = (rt0 + it) * 16;

        v16h af;
        const float* ar = ebase + (size_t)(row0 + n) * kE;
#pragma unroll
        for (int j = 0; j < 8; ++j) {
            af[j]     = (_Float16)ar[k1 + j];
            af[8 + j] = (_Float16)ar[k2 + j];
        }

        v8f c = {};
        c = __builtin_amdgcn_wmma_f32_16x16x32_f16(
                /*neg_a=*/false, af, /*neg_b=*/false, bf,
                /*c_mod=*/(short)0, c, /*reuse_a=*/false, /*reuse_b=*/false);

        // D layout: VGPR j -> M = j + hi*8, N = lane&15
        float* o = obase + (size_t)row0 * kHW;
#pragma unroll
        for (int j = 0; j < 8; ++j)
            o[(size_t)(mb + j) * kHW + n] = c[j];
    }
}

// ============================================================
// Kernel 4: tgt_masks[b,a,hw] = gt_masks[b,hw,unique_idx[b,a]]
//           gt_masks is 884 KB -> fully L2-resident gather.
// ============================================================
__global__ void __launch_bounds__(256)
k4_masks(const float* __restrict__ gt_masks,  // (B,HW,M)
         const int*   __restrict__ uidx,      // (B,A)
         float* __restrict__ out)
{
    const size_t idx = (size_t)blockIdx.x * 256 + threadIdx.x;
    const size_t total = (size_t)kB * kA * kHW;
    if (idx >= total) return;
    const int b  = (int)(idx / ((size_t)kA * kHW));
    const size_t r = idx - (size_t)b * kA * kHW;
    const int a  = (int)(r / kHW);
    const int hw = (int)(r - (size_t)a * kHW);
    const int m  = uidx[b * kA + a];
    const float v = gt_masks[((size_t)b * kHW + hw) * kM + m];
    out[(size_t)b * kOutB + (size_t)kA * (85 + kHW) + (size_t)a * kHW + hw] = v;
}

// ============================================================
extern "C" void kernel_launch(void* const* d_in, const int* in_sizes, int n_in,
                              void* d_out, int out_size, void* d_ws, size_t ws_size,
                              hipStream_t stream)
{
    const float* cls_logits = (const float*)d_in[0];
    const float* pred_boxes = (const float*)d_in[1];
    const float* true_boxes = (const float*)d_in[2];
    const float* mask_embs  = (const float*)d_in[3];
    const float* protos     = (const float*)d_in[4];
    const float* gt_masks   = (const float*)d_in[5];
    const float* anchors    = (const float*)d_in[6];
    const float* scalers    = (const float*)d_in[7];
    const int*   true_cls   = (const int*)d_in[8];
    float* out = (float*)d_out;

    // workspace layout (~2.1 MB)
    float* target   = (float*)d_ws;                          // B*M*A
    float* rowstats = target + (size_t)kB * kM * kA;         // B*M*3
    int*   uidx     = (int*)(rowstats + (size_t)kB * kM * 3);// B*A

    k1_assign<<<kB * kM, 256, 0, stream>>>(cls_logits, pred_boxes, true_boxes,
                                           anchors, true_cls, target, rowstats);

    k2_box<<<(kB * kA + 255) / 256, 256, 0, stream>>>(true_boxes, scalers, true_cls,
                                                      target, rowstats, uidx, out);

    k3_wmma<<<kB * kCT * kSplit, 256, 0, stream>>>(mask_embs, protos, out);

    const size_t nmask = (size_t)kB * kA * kHW;
    k4_masks<<<(unsigned)((nmask + 255) / 256), 256, 0, stream>>>(gt_masks, uidx, out);
}
